// CustomGPT1Model_32667521254277
// MI455X (gfx1250) — matrix operand order repaired and checked by reading, not compiled, not run on hardware
//
#include <hip/hip_runtime.h>

// ---------------------------------------------------------------------------
// CDNA5 (gfx1250, wave32) implementation of the 2-layer custom GPT block.
// All matmuls run through v_wmma_f32_16x16x32_bf16 (f32 accumulate).
// GEMM operand tiles are staged into LDS by the Tensor Data Mover
// (TENSOR_LOAD_TO_LDS + s_wait_tensorcnt) when the builtin is available,
// else by per-lane global_load + ds_store.
// ---------------------------------------------------------------------------

typedef __bf16 bf16;
typedef __attribute__((ext_vector_type(16))) __bf16 v16bf;
typedef __attribute__((ext_vector_type(8)))  __bf16 v8bf;
typedef __attribute__((ext_vector_type(8)))  float  v8f;
typedef __attribute__((ext_vector_type(4)))  unsigned int u32x4;
typedef __attribute__((ext_vector_type(4)))  int          i32x4;
typedef __attribute__((ext_vector_type(8)))  int          i32x8;

#define TM 128
#define TN 128
#define TK 32
#define LDSP 40   // padded LDS row pitch (elements): 64B data + 16B pad

#if defined(__HIP_DEVICE_COMPILE__) && defined(__has_builtin)
#if __has_builtin(__builtin_amdgcn_tensor_load_to_lds)
#define HAVE_TDM 1
#endif
#endif

#if defined(HAVE_TDM)
// Low 32 bits of a generic (flat) LDS pointer = wave-relative LDS byte offset
// (flat LDS aperture: ADDR[63:32]=SHARED_BASE, ADDR[31:0]=LDS offset).
__device__ __forceinline__ unsigned lds_off_u32(const void* p) {
  return (unsigned)(unsigned long long)p;
}

// Issue one TDM 2-D tile load: tile = TM rows x TK bf16 (64B per row),
// row stride = ld elements, LDS pitch = LDSP elements (pad 16B per 64B row).
__device__ __forceinline__ void tdm_load_tile(const bf16* gptr, unsigned lds_addr,
                                              int ld) {
  const unsigned long long ga = (unsigned long long)gptr;
  u32x4 g0;
  g0.x = 1u;                                   // count=1, user descriptor
  g0.y = lds_addr;                             // lds_addr (bytes)
  g0.z = (unsigned)ga;                         // global_addr[31:0]
  g0.w = ((unsigned)(ga >> 32) & 0x1FFFFFFu)   // global_addr[56:32]
       | (2u << 30);                           // type = 2 ("image")
  const int TDIM = 1 << 20;                    // huge tensor dims: no OOB clip
  i32x8 g1;
  g1[0] = (1 << 16)                            // data_size = 1 -> 2 bytes
        | (1 << 20)                            // pad_enable
        | (3 << 22)                            // pad_interval: 16 DW = 64B
        | (3 << 25);                           // pad_amount:    4 DW = 16B
  g1[1] = (TDIM & 0xFFFF) << 16;               // tensor_dim0 lo16 (atomic=0)
  g1[2] = ((TDIM >> 16) & 0xFFFF)              // tensor_dim0 hi16
        | ((TDIM & 0xFFFF) << 16);             // tensor_dim1 lo16
  g1[3] = ((TDIM >> 16) & 0xFFFF)              // tensor_dim1 hi16
        | (TK << 16);                          // tile_dim0 = 32 elements
  g1[4] = TM;                                  // tile_dim1 = 128 rows (dim2=0)
  g1[5] = ld;                                  // tensor_dim0_stride lo32
  g1[6] = 0;                                   // stride0 hi16 | stride1 lo16
  g1[7] = 0;                                   // stride1 hi32
  const i32x4 z4 = {0, 0, 0, 0};
#if __clang_major__ >= 23
  const i32x8 z8 = {0, 0, 0, 0, 0, 0, 0, 0};
  __builtin_amdgcn_tensor_load_to_lds(g0, g1, z4, z4, z8, 0);
#else
  __builtin_amdgcn_tensor_load_to_lds(g0, g1, z4, z4, 0);
#endif
}
#endif  // HAVE_TDM

// ------------------------------ reductions ---------------------------------
__device__ __forceinline__ float wred_sum(float x) {
#pragma unroll
  for (int o = 16; o > 0; o >>= 1) x += __shfl_xor(x, o, 32);
  return x;
}
__device__ __forceinline__ float wred_max(float x) {
#pragma unroll
  for (int o = 16; o > 0; o >>= 1) x = fmaxf(x, __shfl_xor(x, o, 32));
  return x;
}

// ------------------------------ NT GEMM ------------------------------------
// C[M,N] = scale * (A[M,K] * B[N,K]^T)  (+bias[N]) (ReLU) (+res[M,N])
// A, B bf16 row-major. Outputs: f32 (Cf) and/or bf16 (Cb). M,N multiples of
// 128, K multiple of 32. grid = (N/128, M/128), block = 256 (8 waves).
__global__ __launch_bounds__(256)
void gemm_nt(const bf16* __restrict__ A, int lda,
             const bf16* __restrict__ B, int ldb,
             float* __restrict__ Cf, bf16* __restrict__ Cb, int ldc,
             const float* __restrict__ bias, const float* __restrict__ res,
             float scale, int relu, int K) {
  __shared__ bf16 Al[2][TM][LDSP];
  __shared__ bf16 Bl[2][TN][LDSP];

  const int t    = threadIdx.x;
  const int lane = t & 31;
  const int w    = t >> 5;
  const int wm   = w & 3;        // 4 waves along M (32 rows each)
  const int wn   = w >> 2;       // 2 waves along N (64 cols each)
  const int ml   = lane & 15;
  const int half = lane >> 4;
  const int gm0  = blockIdx.y * TM;
  const int gn0  = blockIdx.x * TN;

  v8f acc[2][4];
#pragma unroll
  for (int i = 0; i < 2; ++i)
#pragma unroll
    for (int j = 0; j < 4; ++j) {
      v8f z = {0.f, 0.f, 0.f, 0.f, 0.f, 0.f, 0.f, 0.f};
      acc[i][j] = z;
    }

  const int nk = K / TK;

  // ---- 8 WMMAs per wave per K-step, operands straight from LDS ----
  auto compute = [&](int buf_) {
    v16bf a[2], bo[4];
#pragma unroll
    for (int mi = 0; mi < 2; ++mi) {
      const bf16* ap = &Al[buf_][wm * 32 + mi * 16 + ml][0];
      v8bf lo = *(const v8bf*)(ap + half * 8);        // K = half*8 .. +7
      v8bf hi = *(const v8bf*)(ap + 16 + half * 8);   // K = 16+half*8 .. +7
      a[mi] = __builtin_shufflevector(lo, hi, 0, 1, 2, 3, 4, 5, 6, 7,
                                      8, 9, 10, 11, 12, 13, 14, 15);
    }
#pragma unroll
    for (int ni = 0; ni < 4; ++ni) {
      const bf16* bp = &Bl[buf_][wn * 64 + ni * 16 + ml][0];
      v8bf blo = *(const v8bf*)(bp + half * 8);
      v8bf bhi = *(const v8bf*)(bp + 16 + half * 8);
      bo[ni] = __builtin_shufflevector(blo, bhi, 0, 1, 2, 3, 4, 5, 6, 7,
                                       8, 9, 10, 11, 12, 13, 14, 15);
    }
#pragma unroll
    for (int mi = 0; mi < 2; ++mi)
#pragma unroll
      for (int ni = 0; ni < 4; ++ni)
        acc[mi][ni] = __builtin_amdgcn_wmma_f32_16x16x32_bf16(
            false, a[mi], false, bo[ni], (short)0, acc[mi][ni], false, false);
  };

#if defined(HAVE_TDM)
  // ================= Tensor Data Mover staging =================
  const unsigned ldsA[2] = {lds_off_u32(&Al[0][0][0]), lds_off_u32(&Al[1][0][0])};
  const unsigned ldsB[2] = {lds_off_u32(&Bl[0][0][0]), lds_off_u32(&Bl[1][0][0])};
  const bf16* Abase = A + (size_t)gm0 * lda;
  const bf16* Bbase = B + (size_t)gn0 * ldb;

  if (w == 0) {
    tdm_load_tile(Abase, ldsA[0], lda);
    tdm_load_tile(Bbase, ldsB[0], ldb);
    __builtin_amdgcn_s_wait_tensorcnt(0);
  }
  __syncthreads();

  int buf = 0;
  for (int kt = 0; kt < nk; ++kt) {
    // Issue next tile's DMA: everyone is already past the barrier that ended
    // the last reads of buf^1, so the TDM may overwrite it now.
    if (w == 0 && kt + 1 < nk) {
      tdm_load_tile(Abase + (kt + 1) * TK, ldsA[buf ^ 1], lda);
      tdm_load_tile(Bbase + (kt + 1) * TK, ldsB[buf ^ 1], ldb);
    }
    compute(buf);
    if (w == 0 && kt + 1 < nk) __builtin_amdgcn_s_wait_tensorcnt(0);
    __syncthreads();
    buf ^= 1;
  }
#else
  // ================= fallback: per-lane staging =================
  uint4 ra[2], rb[2];
  auto load_tiles = [&](int kt) {
    const int k0 = kt * TK;
#pragma unroll
    for (int i = 0; i < 2; ++i) {
      const int c  = t + i * 256;     // 512 chunks of 16B per operand tile
      const int r  = c >> 2;
      const int cc = c & 3;
      ra[i] = *(const uint4*)(A + (size_t)(gm0 + r) * lda + k0 + cc * 8);
      rb[i] = *(const uint4*)(B + (size_t)(gn0 + r) * ldb + k0 + cc * 8);
    }
  };
  auto store_tiles = [&](int buf_) {
#pragma unroll
    for (int i = 0; i < 2; ++i) {
      const int c  = t + i * 256;
      const int r  = c >> 2;
      const int cc = c & 3;
      *(uint4*)&Al[buf_][r][cc * 8] = ra[i];
      *(uint4*)&Bl[buf_][r][cc * 8] = rb[i];
    }
  };

  load_tiles(0);
  store_tiles(0);
  __syncthreads();

  int buf = 0;
  for (int kt = 0; kt < nk; ++kt) {
    if (kt + 1 < nk) load_tiles(kt + 1);
    if (kt + 2 < nk) {  // L2 prefetch of the tile after next
      const int c = t, r = c >> 2, cc = c & 3, k2 = (kt + 2) * TK;
      __builtin_prefetch(A + (size_t)(gm0 + r) * lda + k2 + cc * 8, 0, 1);
      __builtin_prefetch(B + (size_t)(gn0 + r) * ldb + k2 + cc * 8, 0, 1);
    }
    compute(buf);
    __syncthreads();
    if (kt + 1 < nk) store_tiles(buf ^ 1);
    __syncthreads();
    buf ^= 1;
  }
#endif

  // ---- epilogue: C layout is M = v + 8*half, N = lane%16 ----
#pragma unroll
  for (int mi = 0; mi < 2; ++mi)
#pragma unroll
    for (int ni = 0; ni < 4; ++ni)
#pragma unroll
      for (int v = 0; v < 8; ++v) {
        const int row = gm0 + wm * 32 + mi * 16 + half * 8 + v;
        const int col = gn0 + wn * 64 + ni * 16 + ml;
        float val = acc[mi][ni][v] * scale;
        if (bias) val += bias[col];
        if (relu) val = fmaxf(val, 0.f);
        if (res)  val += res[(size_t)row * ldc + col];
        if (Cf) Cf[(size_t)row * ldc + col] = val;
        if (Cb) Cb[(size_t)row * ldc + col] = (bf16)val;
      }
}

// --------------------------- masked softmax --------------------------------
// One 256-thread block per row of 4096; writes bf16 probs.
__global__ __launch_bounds__(256)
void softmax_rows(const float* __restrict__ S_, bf16* __restrict__ P,
                  const float* __restrict__ mask, int n) {
  __shared__ float red[8];
  const int row = blockIdx.x;
  const int t   = threadIdx.x;
  const float* sr = S_ + (size_t)row * n;

  float v[16];
  float m = -3.4e38f;
#pragma unroll
  for (int i = 0; i < 16; ++i) {
    const int c = t + i * 256;
    float x = sr[c] + (1.0f - mask[c]) * -1e9f;
    v[i] = x;
    m = fmaxf(m, x);
  }
  m = wred_max(m);
  if ((t & 31) == 0) red[t >> 5] = m;
  __syncthreads();
  float bm = red[0];
#pragma unroll
  for (int k = 1; k < 8; ++k) bm = fmaxf(bm, red[k]);
  __syncthreads();

  float s = 0.f;
#pragma unroll
  for (int i = 0; i < 16; ++i) {
    v[i] = __expf(v[i] - bm);
    s += v[i];
  }
  s = wred_sum(s);
  if ((t & 31) == 0) red[t >> 5] = s;
  __syncthreads();
  float bs = 0.f;
#pragma unroll
  for (int k = 0; k < 8; ++k) bs += red[k];
  const float inv = 1.0f / bs;

#pragma unroll
  for (int i = 0; i < 16; ++i)
    P[(size_t)row * n + t + i * 256] = (bf16)(v[i] * inv);
}

// ------------------------------ layernorm ----------------------------------
// One 256-thread block per row (D=1024). Emits bf16 row; optionally a
// transposed bf16 copy NT[b][d][s] for the probs@n GEMM.
__global__ __launch_bounds__(256)
void ln_rows(const float* __restrict__ X, bf16* __restrict__ N_,
             bf16* __restrict__ NT, const float* __restrict__ w,
             const float* __restrict__ b_, int D_, int S_) {
  __shared__ float red[8];
  const int row = blockIdx.x;
  const int t   = threadIdx.x;
  const float* xr = X + (size_t)row * D_;

  float v[4];
  float s = 0.f;
#pragma unroll
  for (int i = 0; i < 4; ++i) {
    v[i] = xr[t + i * 256];
    s += v[i];
  }
  s = wred_sum(s);
  if ((t & 31) == 0) red[t >> 5] = s;
  __syncthreads();
  float tot = 0.f;
#pragma unroll
  for (int k = 0; k < 8; ++k) tot += red[k];
  const float mean = tot / (float)D_;
  __syncthreads();

  float var = 0.f;
#pragma unroll
  for (int i = 0; i < 4; ++i) {
    const float d = v[i] - mean;
    var += d * d;
  }
  var = wred_sum(var);
  if ((t & 31) == 0) red[t >> 5] = var;
  __syncthreads();
  float vt = 0.f;
#pragma unroll
  for (int k = 0; k < 8; ++k) vt += red[k];
  const float r = rsqrtf(vt / (float)D_ + 1e-5f);

  const int bb = row / S_;
  const int ss = row % S_;
#pragma unroll
  for (int i = 0; i < 4; ++i) {
    const int d = t + i * 256;
    const float y = (v[i] - mean) * r * w[d] + b_[d];
    N_[(size_t)row * D_ + d] = (bf16)y;
    if (NT) NT[((size_t)bb * D_ + d) * S_ + ss] = (bf16)y;
  }
}

// ------------------------------ embeddings ---------------------------------
__global__ __launch_bounds__(256)
void embed_k(const int* __restrict__ ids, const float* __restrict__ tok,
             const float* __restrict__ pos, const float* __restrict__ side,
             float* __restrict__ X, int S_, int D_) {
  const int row = blockIdx.x;
  const int bb = row / S_, ss = row % S_;
  const int id = ids[row];
  const int t = threadIdx.x;
#pragma unroll
  for (int i = 0; i < 4; ++i) {
    const int d = t + i * 256;
    X[(size_t)row * D_ + d] =
        tok[(size_t)id * D_ + d] + pos[(size_t)ss * D_ + d] + side[bb * D_ + d];
  }
}

// side[b][d] = concat(occ_emb[occ[b]], gen_emb[gen[b]]) @ proj_W + proj_b
__global__ __launch_bounds__(256)
void side_k(const int* __restrict__ occ_id, const int* __restrict__ gen_id,
            const float* __restrict__ occ_e, const float* __restrict__ gen_e,
            const float* __restrict__ pW, const float* __restrict__ pb,
            float* __restrict__ side) {
  __shared__ float agg[72];
  const int bb = blockIdx.x;
  const int t = threadIdx.x;
  if (t < 64) agg[t] = occ_e[occ_id[bb] * 64 + t];
  else if (t < 72) agg[t] = gen_e[gen_id[bb] * 8 + (t - 64)];
  __syncthreads();
#pragma unroll
  for (int i = 0; i < 4; ++i) {
    const int d = t + i * 256;
    float s = pb[d];
    for (int k = 0; k < 72; ++k) s += agg[k] * pW[k * 1024 + d];
    side[bb * 1024 + d] = s;
  }
}

// ---------------------- transpose + f32 -> bf16 cast -----------------------
// in f32 [R,C] -> out bf16 [C,R].  grid = (C/32, R/32), block = 256.
__global__ __launch_bounds__(256)
void tcast_k(const float* __restrict__ in, bf16* __restrict__ out, int R, int C) {
  __shared__ float tile[32][33];
  const int tx = threadIdx.x & 31;
  const int ty = threadIdx.x >> 5;   // 0..7
  const int c0 = blockIdx.x * 32;
  const int r0 = blockIdx.y * 32;
#pragma unroll
  for (int j = 0; j < 4; ++j)
    tile[ty + j * 8][tx] = in[(size_t)(r0 + ty + j * 8) * C + c0 + tx];
  __syncthreads();
#pragma unroll
  for (int j = 0; j < 4; ++j)
    out[(size_t)(c0 + ty + j * 8) * R + r0 + tx] = (bf16)tile[tx][ty + j * 8];
}

// ------------------------------ host driver --------------------------------
extern "C" void kernel_launch(void* const* d_in, const int* in_sizes, int n_in,
                              void* d_out, int out_size, void* d_ws,
                              size_t ws_size, hipStream_t stream) {
  (void)in_sizes; (void)n_in; (void)out_size; (void)ws_size;
  const int Bc = 4, Sc = 4096, Dc = 1024, FFc = 2048, Vc = 512;
  const size_t BS = (size_t)Bc * Sc;

  const int*   input_ids = (const int*)d_in[0];
  const int*   occ_ids   = (const int*)d_in[1];
  const int*   gen_ids   = (const int*)d_in[2];
  const float* amask     = (const float*)d_in[3];
  const float* tok       = (const float*)d_in[4];
  const float* pos       = (const float*)d_in[5];
  const float* occ_e     = (const float*)d_in[6];
  const float* gen_e     = (const float*)d_in[7];
  const float* projW     = (const float*)d_in[8];
  const float* projb     = (const float*)d_in[9];
  const float* lnw       = (const float*)d_in[10];
  const float* lnb       = (const float*)d_in[11];
  const float* w1        = (const float*)d_in[12];
  const float* b1        = (const float*)d_in[13];
  const float* w2        = (const float*)d_in[14];
  const float* b2        = (const float*)d_in[15];
  const float* oW        = (const float*)d_in[16];
  const float* ob        = (const float*)d_in[17];

  char* ws = (char*)d_ws;
  size_t off = 0;
  auto alloc = [&](size_t bytes) -> void* {
    void* p = ws + off;
    off += (bytes + 255) & ~(size_t)255;
    return p;
  };
  float* x      = (float*)alloc(BS * Dc * 4);            // activations (f32)
  float* attn   = (float*)alloc(BS * Dc * 4);            // attn output (f32)
  bf16*  nb     = (bf16*) alloc(BS * Dc * 2);            // LN out (bf16)
  bf16*  nTb    = (bf16*) alloc(BS * Dc * 2);            // LN out transposed
  bf16*  ff1    = (bf16*) alloc(BS * FFc * 2);           // FFN hidden (bf16)
  bf16*  xb     = (bf16*) alloc(BS * Dc * 2);            // x as bf16 (final)
  float* scores = (float*)alloc((size_t)Sc * Sc * 4);    // one batch of scores
  bf16*  probs  = (bf16*) alloc((size_t)Sc * Sc * 2);    // one batch of probs
  bf16*  W1T    = (bf16*) alloc((size_t)2 * FFc * Dc * 2);
  bf16*  W2T    = (bf16*) alloc((size_t)2 * Dc * FFc * 2);
  bf16*  oWT    = (bf16*) alloc((size_t)Vc * Dc * 2);
  float* side   = (float*)alloc((size_t)Bc * Dc * 4);

  // side embeddings + token/pos embedding
  side_k<<<Bc, 256, 0, stream>>>(occ_ids, gen_ids, occ_e, gen_e, projW, projb, side);
  embed_k<<<(int)BS, 256, 0, stream>>>(input_ids, tok, pos, side, x, Sc, Dc);

  // pre-transpose weights to bf16 [N,K] form
  for (int l = 0; l < 2; ++l) {
    tcast_k<<<dim3(FFc / 32, Dc / 32), 256, 0, stream>>>(
        w1 + (size_t)l * Dc * FFc, W1T + (size_t)l * FFc * Dc, Dc, FFc);
    tcast_k<<<dim3(Dc / 32, FFc / 32), 256, 0, stream>>>(
        w2 + (size_t)l * FFc * Dc, W2T + (size_t)l * Dc * FFc, FFc, Dc);
  }
  tcast_k<<<dim3(Vc / 32, Dc / 32), 256, 0, stream>>>(oW, oWT, Dc, Vc);

  const float qk_scale = 1.0f / 32.0f;  // 1/sqrt(D), D=1024

  for (int l = 0; l < 2; ++l) {
    // n = LN(x)  (row-major + transposed bf16 copies)
    ln_rows<<<(int)BS, 256, 0, stream>>>(x, nb, nTb, lnw + l * Dc, lnb + l * Dc, Dc, Sc);

    for (int b = 0; b < Bc; ++b) {
      const bf16* nb_b  = nb  + (size_t)b * Sc * Dc;
      const bf16* nTb_b = nTb + (size_t)b * Dc * Sc;
      // scores = scale * n n^T
      gemm_nt<<<dim3(Sc / TN, Sc / TM), 256, 0, stream>>>(
          nb_b, Dc, nb_b, Dc, scores, nullptr, Sc,
          nullptr, nullptr, qk_scale, 0, Dc);
      // probs = softmax(scores + mask)
      softmax_rows<<<Sc, 256, 0, stream>>>(scores, probs, amask + (size_t)b * Sc, Sc);
      // attn = probs @ n + x
      gemm_nt<<<dim3(Dc / TN, Sc / TM), 256, 0, stream>>>(
          probs, Sc, nTb_b, Sc, attn + (size_t)b * Sc * Dc, nullptr, Dc,
          nullptr, x + (size_t)b * Sc * Dc, 1.0f, 0, Sc);
    }

    // na = LN(attn) (reuse nb buffer; no transpose needed)
    ln_rows<<<(int)BS, 256, 0, stream>>>(attn, nb, nullptr, lnw + l * Dc, lnb + l * Dc, Dc, Sc);

    // ff1 = relu(na @ W1 + b1)   -> bf16
    gemm_nt<<<dim3(FFc / TN, (int)BS / TM), 256, 0, stream>>>(
        nb, Dc, W1T + (size_t)l * FFc * Dc, Dc, nullptr, ff1, FFc,
        b1 + (size_t)l * FFc, nullptr, 1.0f, 1, Dc);

    // x = ff1 @ W2 + b2 + attn   (last layer also emits bf16 copy)
    gemm_nt<<<dim3(Dc / TN, (int)BS / TM), 256, 0, stream>>>(
        ff1, FFc, W2T + (size_t)l * Dc * FFc, FFc, x, (l == 1) ? xb : nullptr, Dc,
        b2 + (size_t)l * Dc, attn, 1.0f, 0, FFc);
  }

  // logits = x @ out_W + out_b  -> d_out (f32)
  gemm_nt<<<dim3(Vc / TN, (int)BS / TM), 256, 0, stream>>>(
      xb, Dc, oWT, Dc, (float*)d_out, nullptr, Vc, ob, nullptr, 1.0f, 0, Dc);
}
